// RNNBase_22119081574733
// MI455X (gfx1250) — matrix operand order repaired and checked by reading, not compiled, not run on hardware
//
#include <hip/hip_runtime.h>
#include <hip/hip_bf16.h>

typedef __attribute__((ext_vector_type(16))) __bf16 v16bf;
typedef __attribute__((ext_vector_type(8)))  float  v8f;
typedef unsigned short u16;
typedef unsigned int   u32;
typedef unsigned long long u64;

// ---------- helpers ----------
static __device__ __forceinline__ u16 f2bf(float f) {
    u32 u = __builtin_bit_cast(u32, f);
    u = (u + 0x7FFFu + ((u >> 16) & 1u)) >> 16;     // RNE
    return (u16)u;
}
static __device__ __forceinline__ __bf16 bfbits(u16 b) {
    return __builtin_bit_cast(__bf16, b);
}
static __device__ __forceinline__ float sigmoidf(float x) {
    return 1.0f / (1.0f + __expf(-x));
}
// CDNA5 async copy: global -> LDS, 16B per lane, tracked by ASYNCcnt (bypasses VGPRs)
static __device__ __forceinline__ void async_copy_b128(u32 lds_byte_addr, const void* gptr) {
    asm volatile("global_load_async_to_lds_b128 %0, %1, off"
                 :: "v"(lds_byte_addr), "v"((u64)(uintptr_t)gptr) : "memory");
}
static __device__ __forceinline__ void wait_async0() {
    asm volatile("s_wait_asynccnt 0" ::: "memory");
}

// ---------- weight transpose + f32->bf16 : [1024][256] -> [256][1024] ----------
__global__ void convert_w(const float* __restrict__ in, u16* __restrict__ out) {
    int idx = blockIdx.x * 256 + threadIdx.x;       // 262144 total
    int n = idx >> 8, k = idx & 255;
    out[k * 1024 + n] = f2bf(in[idx]);
}

// ---------- elementwise f32 -> bf16 ----------
__global__ void convert_x(const float* __restrict__ in, u16* __restrict__ out, int n) {
    int idx = blockIdx.x * 256 + threadIdx.x;
    if (idx < n) out[idx] = f2bf(in[idx]);
}

// ---------- big gate GEMM: C[M,1024] = A_bf16[M,256] * B_bf16[256,1024] + bx + bh ----------
// block tile 128x128, 256 threads (8 waves as 4Mw x 2Nw, each wave 32x64)
#define ASTR 40   // A LDS row stride in ushorts: 80B rows -> 16B aligned + conflict-free
__global__ __launch_bounds__(256) void gemm_gates(
    const u16* __restrict__ A, const u16* __restrict__ Bw,
    const float* __restrict__ bx, const float* __restrict__ bh,
    float* __restrict__ Cout)
{
    __shared__ u16 As[128 * ASTR];      // [row][k0..31], async-copied row-major
    __shared__ u16 Bs[128 * 34];        // transposed: [col][k0..31], stride 34

    const int tid  = threadIdx.x;
    const int lane = tid & 31;
    const int wv   = tid >> 5;
    const int mw   = wv >> 1;           // 0..3
    const int nw   = wv & 1;            // 0..1
    const size_t m0 = (size_t)blockIdx.x * 128;
    const int    n0 = blockIdx.y * 128;

    u32* As32 = (u32*)As;
    u32* Bs32 = (u32*)Bs;
    const u32 AsBase = (u32)(uintptr_t)&As[0];

    v8f C[2][4];
#pragma unroll
    for (int mt = 0; mt < 2; mt++)
#pragma unroll
        for (int nt = 0; nt < 4; nt++)
#pragma unroll
            for (int i = 0; i < 8; i++) C[mt][nt][i] = 0.0f;

    for (int kb = 0; kb < 8; kb++) {
        // stage A tile 128x32 bf16 via async global->LDS b128 (2 x 16B chunks/thread)
#pragma unroll
        for (int s = 0; s < 2; s++) {
            int q = tid + 256 * s;                    // 512 chunks of 16B
            int r = q >> 2, cpos = q & 3;
            async_copy_b128(AsBase + (u32)(r * (ASTR * 2) + cpos * 16),
                            A + (m0 + r) * 256 + kb * 32 + cpos * 8);
        }
        // stage B tile 32x128 bf16, transposed into Bs[col][k] (register path: needs transpose)
        {
            int k = tid >> 3, c16 = (tid & 7) * 16;
            const uint4* p = (const uint4*)(Bw + (size_t)(kb * 32 + k) * 1024 + n0 + c16);
            u32 v[8];
            *(uint4*)&v[0] = p[0];
            *(uint4*)&v[4] = p[1];
#pragma unroll
            for (int j = 0; j < 8; j++) {
                Bs[(c16 + 2 * j) * 34 + k]     = (u16)(v[j] & 0xFFFF);
                Bs[(c16 + 2 * j + 1) * 34 + k] = (u16)(v[j] >> 16);
            }
        }
        wait_async0();                  // our async copies landed in LDS
        __syncthreads();                // everyone else's too

        // fragments + WMMA
        v16bf a[2], b[4];
        const int khalf = (lane & 16) >> 1;     // +8 for upper lane half (A layout)
#pragma unroll
        for (int mt = 0; mt < 2; mt++) {
            int row = mw * 32 + mt * 16 + (lane & 15);
#pragma unroll
            for (int j = 0; j < 8; j++) {
                int kk = (j & 3) * 2 + ((j >> 2) * 16) + khalf;
                u32 d = As32[row * (ASTR / 2) + (kk >> 1)];
                a[mt][2 * j]     = bfbits((u16)(d & 0xFFFF));
                a[mt][2 * j + 1] = bfbits((u16)(d >> 16));
            }
        }
#pragma unroll
        for (int nt = 0; nt < 4; nt++) {
            int col = nw * 64 + nt * 16 + (lane & 15);
#pragma unroll
            for (int j = 0; j < 8; j++) {
                int kk = (j & 3) * 2 + ((j >> 2) * 16) + ((lane & 16) >> 1);
                u32 d = Bs32[col * 17 + (kk >> 1)];
                b[nt][2 * j]     = bfbits((u16)(d & 0xFFFF));
                b[nt][2 * j + 1] = bfbits((u16)(d >> 16));
            }
        }
#pragma unroll
        for (int mt = 0; mt < 2; mt++)
#pragma unroll
            for (int nt = 0; nt < 4; nt++)
                C[mt][nt] = __builtin_amdgcn_wmma_f32_16x16x32_bf16(
                    false, a[mt], false, b[nt], (short)0, C[mt][nt], false, false);
        __syncthreads();
    }

    // epilogue: add both biases, store f32
#pragma unroll
    for (int nt = 0; nt < 4; nt++) {
        int gcol = n0 + nw * 64 + nt * 16 + (lane & 15);
        float bias = bx[gcol] + bh[gcol];
#pragma unroll
        for (int mt = 0; mt < 2; mt++) {
            size_t grow = m0 + mw * 32 + mt * 16 + ((lane & 16) >> 1);
#pragma unroll
            for (int i = 0; i < 8; i++)
                Cout[(grow + i) * 1024 + gcol] = C[mt][nt][i] + bias;
        }
    }
}

// ---------- persistent recurrent scan: one workgroup, 32 waves, Whh pinned in VGPRs ----------
__global__ __launch_bounds__(1024, 1) void lstm_scan(
    const float* __restrict__ gx,     // [T][64][1024], biases folded in
    const u16*   __restrict__ WhT,    // [256][1024] bf16
    const float* __restrict__ h0,     // [64][256]
    const float* __restrict__ c0,     // [64][256]
    u16*   __restrict__ outB,         // bf16 [T][64][256] or null
    float* __restrict__ outF,         // f32  [T][64][256] or null
    float* __restrict__ hN, float* __restrict__ cN, int T)
{
    __shared__ u16   hsm[64 * 258];       // h (bf16), odd-dword row stride
    __shared__ float gsm[64 * 1025];      // gate pre-activations (f32)

    const int tid  = threadIdx.x;
    const int lane = tid & 31;
    const int w    = tid >> 5;            // wave id: owns gate cols [32w, 32w+32)

    // pin B (WhhT) tiles in registers: 2 N-tiles x 8 K-tiles, 128 VGPRs/lane
    v16bf Breg[2][8];
#pragma unroll
    for (int nt = 0; nt < 2; nt++) {
        int col = w * 32 + nt * 16 + (lane & 15);
#pragma unroll
        for (int kb = 0; kb < 8; kb++)
#pragma unroll
            for (int j = 0; j < 8; j++) {
                int kk = kb * 32 + (j & 3) * 2 + ((j >> 2) * 16) + ((lane & 16) >> 1);
                Breg[nt][kb][2 * j]     = bfbits(WhT[(size_t)kk * 1024 + col]);
                Breg[nt][kb][2 * j + 1] = bfbits(WhT[(size_t)(kk + 1) * 1024 + col]);
            }
    }

    // private cell state (16 elements/thread) + h init
    float creg[16];
#pragma unroll
    for (int r = 0; r < 16; r++) {
        int idx = tid * 16 + r;
        creg[r] = c0[idx];
        hsm[(idx >> 8) * 258 + (idx & 255)] = f2bf(h0[idx]);
    }
    __syncthreads();

    for (int t = 0; t < T; t++) {
        const float* gxt = gx + (size_t)t * 65536;
        if (t + 1 < T) __builtin_prefetch(gx + (size_t)(t + 1) * 65536, 0, 1);

        // two M-passes (rows 0..31, 32..63) to bound accumulator VGPRs
#pragma unroll
        for (int mp = 0; mp < 2; mp++) {
            v8f Cc[2][2];
#pragma unroll
            for (int mt = 0; mt < 2; mt++) {
                int row0 = (mp * 2 + mt) * 16 + ((lane & 16) >> 1);
#pragma unroll
                for (int nt = 0; nt < 2; nt++) {
                    int col = w * 32 + nt * 16 + (lane & 15);
#pragma unroll
                    for (int i = 0; i < 8; i++)
                        Cc[mt][nt][i] = gxt[(size_t)(row0 + i) * 1024 + col];
                }
            }
#pragma unroll
            for (int kb = 0; kb < 8; kb++) {
                v16bf a[2];
                const int khalf = (lane & 16) >> 1;
#pragma unroll
                for (int mt = 0; mt < 2; mt++) {
                    int row = (mp * 2 + mt) * 16 + (lane & 15);
#pragma unroll
                    for (int j = 0; j < 8; j++) {
                        int kk = kb * 32 + (j & 3) * 2 + ((j >> 2) * 16) + khalf;
                        u32 d = *(const u32*)&hsm[row * 258 + kk];
                        a[mt][2 * j]     = bfbits((u16)(d & 0xFFFF));
                        a[mt][2 * j + 1] = bfbits((u16)(d >> 16));
                    }
                }
#pragma unroll
                for (int mt = 0; mt < 2; mt++)
#pragma unroll
                    for (int nt = 0; nt < 2; nt++)
                        Cc[mt][nt] = __builtin_amdgcn_wmma_f32_16x16x32_bf16(
                            false, a[mt], false, Breg[nt][kb], (short)0,
                            Cc[mt][nt], false, false);
            }
            // spill gates to LDS
#pragma unroll
            for (int mt = 0; mt < 2; mt++) {
                int row0 = (mp * 2 + mt) * 16 + ((lane & 16) >> 1);
#pragma unroll
                for (int nt = 0; nt < 2; nt++) {
                    int col = w * 32 + nt * 16 + (lane & 15);
#pragma unroll
                    for (int i = 0; i < 8; i++)
                        gsm[(row0 + i) * 1025 + col] = Cc[mt][nt][i];
                }
            }
        }
        __syncthreads();

        // elementwise LSTM cell update (all 1024 threads, 16 elems each)
#pragma unroll
        for (int r = 0; r < 16; r++) {
            int idx = tid * 16 + r;
            int b = idx >> 8, n = idx & 255;
            float gi = gsm[b * 1025 + n];
            float gf = gsm[b * 1025 + 256 + n];
            float gc = gsm[b * 1025 + 512 + n];
            float go = gsm[b * 1025 + 768 + n];
            float ig = sigmoidf(gi);
            float fg = sigmoidf(gf);
            float cg = tanhf(gc);
            float og = sigmoidf(go);
            float cc = fg * creg[r] + ig * cg;
            creg[r] = cc;
            float hh = og * tanhf(cc);
            hsm[b * 258 + n] = f2bf(hh);
            if (outF) outF[(size_t)t * 16384 + idx] = hh;
            if (outB) outB[(size_t)t * 16384 + idx] = f2bf(hh);
            if (t == T - 1) hN[idx] = hh;
        }
        __syncthreads();
    }

#pragma unroll
    for (int r = 0; r < 16; r++) cN[tid * 16 + r] = creg[r];
}

// ---------- host ----------
extern "C" void kernel_launch(void* const* d_in, const int* in_sizes, int n_in,
                              void* d_out, int out_size, void* d_ws, size_t ws_size,
                              hipStream_t stream) {
    (void)in_sizes; (void)n_in; (void)out_size; (void)ws_size;
    const float* x    = (const float*)d_in[0];
    const float* h0   = (const float*)d_in[1];
    const float* c0   = (const float*)d_in[2];
    const float* Wxh0 = (const float*)d_in[3];
    const float* bxh0 = (const float*)d_in[4];
    const float* Whh0 = (const float*)d_in[5];
    const float* bhh0 = (const float*)d_in[6];
    const float* Wxh1 = (const float*)d_in[7];
    const float* bxh1 = (const float*)d_in[8];
    const float* Whh1 = (const float*)d_in[9];
    const float* bhh1 = (const float*)d_in[10];
    float* out = (float*)d_out;

    // workspace carve-up
    u16* WT0   = (u16*)d_ws;                  // 256K elems each
    u16* WhT0  = WT0   + 262144;
    u16* WT1   = WhT0  + 262144;
    u16* WhT1  = WT1   + 262144;
    u16* xb    = WhT1  + 262144;              // 33554432 bf16
    u16* hs0b  = xb    + 33554432;            // 33554432 bf16
    float* gx  = (float*)(hs0b + 33554432);   // 134217728 f32 (reused per layer)

    const int NX = 2048 * 64 * 256;           // 33554432
    const int OUT = NX;                       // f32 output length

    convert_w<<<dim3(1024), dim3(256), 0, stream>>>(Wxh0, WT0);
    convert_w<<<dim3(1024), dim3(256), 0, stream>>>(Whh0, WhT0);
    convert_w<<<dim3(1024), dim3(256), 0, stream>>>(Wxh1, WT1);
    convert_w<<<dim3(1024), dim3(256), 0, stream>>>(Whh1, WhT1);
    convert_x<<<dim3(NX / 256), dim3(256), 0, stream>>>(x, xb, NX);

    // layer 0
    gemm_gates<<<dim3(1024, 8), dim3(256), 0, stream>>>(xb, WT0, bxh0, bhh0, gx);
    lstm_scan<<<dim3(1), dim3(1024), 0, stream>>>(
        gx, WhT0, h0, c0, hs0b, nullptr,
        out + OUT,            // hT0
        out + OUT + 32768,    // cT0
        2048);

    // layer 1
    gemm_gates<<<dim3(1024, 8), dim3(256), 0, stream>>>(hs0b, WT1, bxh1, bhh1, gx);
    lstm_scan<<<dim3(1), dim3(1024), 0, stream>>>(
        gx, WhT1, h0 + 16384, c0 + 16384, nullptr, out,
        out + OUT + 16384,    // hT1
        out + OUT + 49152,    // cT1
        2048);
}
